// StartPointAttnModelFull_90074054131991
// MI455X (gfx1250) — matrix operand order repaired and checked by reading, not compiled, not run on hardware
//
#include <hip/hip_runtime.h>
#include <hip/hip_bf16.h>

// ---------------------------------------------------------------------------
// Model dims
// ---------------------------------------------------------------------------
#define TT   512
#define BB   128
#define DIN  1024
#define HH   128
#define G4   512          // 4*H
#define VV   4
#define NHD  4            // heads
#define HD   32           // head dim
#define MAXL 64
#define K0CAT 288         // [y(4)|ct(128)|pad(28)|h0(128)]
#define K1CAT 256         // [h0|h1]

typedef __attribute__((ext_vector_type(16))) __bf16 v16bf;
typedef __attribute__((ext_vector_type(8)))  float  v8f;

union FragU { v16bf v; uint4 q[2]; };

// ---------------------------------------------------------------------------
// WMMA fragment loaders (CDNA5 ISA 7.12.2 layouts, wave32)
// A-matrix 16x32 bf16: lane<16 row=lane holds K={k0..k0+7, k0+16..k0+23}
//                      lane>=16 row=lane-16 holds K={k0+8..k0+15, k0+24..k0+31}
// ---------------------------------------------------------------------------
__device__ __forceinline__ v16bf load_fragA(const __bf16* base, int ld, int row0, int k0) {
    const int lane = threadIdx.x & 31;
    const int r    = row0 + (lane & 15);
    const int kk   = k0 + ((lane >> 4) << 3);
    const __bf16* p = base + (size_t)r * ld + kk;
    FragU u;
    u.q[0] = *(const uint4*)(p);
    u.q[1] = *(const uint4*)(p + 16);
    return u.v;
}

// B-matrix 32x16 bf16 (W[N,K] rows fed as B columns):
// lane<16 col=lane holds K={k0..k0+7, k0+16..k0+23} ; lane>=16 col=lane-16 holds K={k0+8..k0+15, k0+24..k0+31}
__device__ __forceinline__ v16bf load_fragB(const __bf16* base, int ld, int row0, int k0) {
    const int lane = threadIdx.x & 31;
    const int r    = row0 + (lane & 15);
    const int kk   = k0 + ((lane >> 4) << 3);
    const __bf16* p = base + (size_t)r * ld + kk;
    FragU u;
    u.q[0] = *(const uint4*)(p);
    u.q[1] = *(const uint4*)(p + 16);
    return u.v;
}

__device__ __forceinline__ v8f wmma_bf16(v16bf a, v16bf b, v8f c) {
    return __builtin_amdgcn_wmma_f32_16x16x32_bf16(false, a, false, b, (short)0, c, false, false);
}

__device__ __forceinline__ float sigmoidf_(float x) { return 1.f / (1.f + __expf(-x)); }

// ---------------------------------------------------------------------------
// Elementwise prep kernels
// ---------------------------------------------------------------------------
__global__ void f32_to_bf16_k(const float* __restrict__ s, __bf16* __restrict__ d, long long n) {
    long long i = (long long)blockIdx.x * blockDim.x + threadIdx.x;
    long long st = (long long)gridDim.x * blockDim.x;
    for (; i < n; i += st) d[i] = (__bf16)s[i];
}

// wcat0[512][288] = [Wih0(132) | zeros(28) | Whh0(128)]
__global__ void build_wcat0_k(const float* __restrict__ Wih, const float* __restrict__ Whh,
                              __bf16* __restrict__ out) {
    long long i = (long long)blockIdx.x * blockDim.x + threadIdx.x;
    long long st = (long long)gridDim.x * blockDim.x;
    for (; i < (long long)G4 * K0CAT; i += st) {
        int r = (int)(i / K0CAT), c = (int)(i % K0CAT);
        float v = 0.f;
        if (c < 132)       v = Wih[r * 132 + c];
        else if (c >= 160) v = Whh[r * 128 + (c - 160)];
        out[i] = (__bf16)v;
    }
}

// wcat1[512][256] = [Wih1(128) | Whh1(128)]
__global__ void build_wcat1_k(const float* __restrict__ Wih, const float* __restrict__ Whh,
                              __bf16* __restrict__ out) {
    long long i = (long long)blockIdx.x * blockDim.x + threadIdx.x;
    long long st = (long long)gridDim.x * blockDim.x;
    for (; i < (long long)G4 * K1CAT; i += st) {
        int r = (int)(i / K1CAT), c = (int)(i % K1CAT);
        float v = (c < 128) ? Wih[r * 128 + c] : Whh[r * 128 + (c - 128)];
        out[i] = (__bf16)v;
    }
}

// ---------------------------------------------------------------------------
// Grid WMMA GEMM: C[M,N](f32) = A_bf16[M,K] @ W_bf16[N,K]^T + bias[N]
// block = 256 thr (8 waves), block tile 64(M) x 128(N), each wave 32x32
// ---------------------------------------------------------------------------
__global__ void wmma_gemm_k(const __bf16* __restrict__ A, int lda,
                            const __bf16* __restrict__ W, int ldw,
                            const float* __restrict__ bias,
                            float* __restrict__ C, int ldc, int K) {
    const int lane = threadIdx.x & 31;
    const int wave = threadIdx.x >> 5;
    const int m0 = blockIdx.x * 64  + (wave & 1) * 32;
    const int n0 = blockIdx.y * 128 + (wave >> 1) * 32;
    v8f acc00 = {}, acc01 = {}, acc10 = {}, acc11 = {};
    for (int k0 = 0; k0 < K; k0 += 32) {
        v16bf a0 = load_fragA(A, lda, m0,      k0);
        v16bf a1 = load_fragA(A, lda, m0 + 16, k0);
        v16bf b0 = load_fragB(W, ldw, n0,      k0);
        v16bf b1 = load_fragB(W, ldw, n0 + 16, k0);
        acc00 = wmma_bf16(a0, b0, acc00);
        acc01 = wmma_bf16(a0, b1, acc01);
        acc10 = wmma_bf16(a1, b0, acc10);
        acc11 = wmma_bf16(a1, b1, acc11);
    }
    const int cl = lane & 15, rb = (lane >> 4) << 3;
#pragma unroll
    for (int mi = 0; mi < 2; ++mi) {
        v8f aL = mi ? acc10 : acc00;
        v8f aR = mi ? acc11 : acc01;
        int mrow = m0 + mi * 16 + rb;
        float bL = bias[n0 + cl];
        float bR = bias[n0 + 16 + cl];
#pragma unroll
        for (int i = 0; i < 8; ++i) {
            C[(size_t)(mrow + i) * ldc + n0 + cl]      = aL[i] + bL;
            C[(size_t)(mrow + i) * ldc + n0 + 16 + cl] = aR[i] + bR;
        }
    }
}

// ---------------------------------------------------------------------------
// Persistent LSTM scan: one workgroup (512 thr = 16 waves), T sequential steps.
// LDS: h bf16 [128][136] + gates f32 [128][516]  (~292KB of the 320KB WGP LDS)
// Whh B-fragments are preloaded into VGPRs ONCE (wave w owns N-stripe
// nt in {2w, 2w+1}: 8 frags = 64 VGPRs), so the recurrent GEMM touches only
// LDS + registers. pre[t] is added in the gating phase (contiguous loads) and
// pre[t+1] is prefetched (global_prefetch_b8) during the WMMA phase.
// ---------------------------------------------------------------------------
#define HLDS 136
#define GLDS 516
__global__ void lstm_scan_k(const float* __restrict__ pre,   // [T,B,512]
                            const __bf16* __restrict__ Whh,  // [512,128]
                            __bf16* __restrict__ hs,         // [T,B,128]
                            int T) {
    extern __shared__ char smem[];
    __bf16* h_lds = (__bf16*)smem;                       // 128*136*2
    float*  g_lds = (float*)(smem + 128 * HLDS * 2);     // 128*516*4
    const int tid  = threadIdx.x;
    const int lane = tid & 31;
    const int wave = tid >> 5;

    for (int i = tid; i < 128 * HLDS; i += 512) h_lds[i] = (__bf16)0.f;
    float c[32];
#pragma unroll
    for (int j = 0; j < 32; ++j) c[j] = 0.f;
    const int gb  = tid >> 2;
    const int gu0 = (tid & 3) * 32;

    // hold all recurrent-weight fragments in registers for the whole scan
    v16bf bf[2][4];
#pragma unroll
    for (int nn = 0; nn < 2; ++nn)
#pragma unroll
        for (int kt = 0; kt < 4; ++kt)
            bf[nn][kt] = load_fragB(Whh, HH, (wave * 2 + nn) * 16, kt * 32);
    __syncthreads();

    for (int t = 0; t < T; ++t) {
        // prefetch next timestep's pre-activations (256KB, 4 lines/thread)
        if (t + 1 < T) {
            const float* pn = pre + (size_t)(t + 1) * (BB * G4);
#pragma unroll
            for (int i = 0; i < 4; ++i)
                __builtin_prefetch(pn + (size_t)(tid + i * 512) * 32, 0, 1);
        }
        // g = h @ Whh^T : wave w covers n-cols [32w, 32w+32), all 128 rows
        for (int mt = 0; mt < 8; ++mt) {
            v16bf a[4];
#pragma unroll
            for (int kt = 0; kt < 4; ++kt)
                a[kt] = load_fragA(h_lds, HLDS, mt * 16, kt * 32);
#pragma unroll
            for (int nn = 0; nn < 2; ++nn) {
                v8f acc = {};
#pragma unroll
                for (int kt = 0; kt < 4; ++kt)
                    acc = wmma_bf16(a[kt], bf[nn][kt], acc);
                const int col = (wave * 2 + nn) * 16 + (lane & 15);
                const int rb  = mt * 16 + ((lane >> 4) << 3);
#pragma unroll
                for (int i = 0; i < 8; ++i)
                    g_lds[(rb + i) * GLDS + col] = acc[i];
            }
        }
        __syncthreads();
        // gating: add pre[t] (contiguous 32-float runs per thread) + activations
        const float* pre_t = pre + (size_t)t * (BB * G4);
#pragma unroll
        for (int j = 0; j < 32; ++j) {
            const int u = gu0 + j;
            float gi = g_lds[gb * GLDS + u]       + pre_t[gb * G4 + u];
            float gf = g_lds[gb * GLDS + u + 128] + pre_t[gb * G4 + u + 128];
            float gg = g_lds[gb * GLDS + u + 256] + pre_t[gb * G4 + u + 256];
            float go = g_lds[gb * GLDS + u + 384] + pre_t[gb * G4 + u + 384];
            c[j] = sigmoidf_(gf) * c[j] + sigmoidf_(gi) * tanhf(gg);
            float h = sigmoidf_(go) * tanhf(c[j]);
            h_lds[gb * HLDS + u] = (__bf16)h;
            hs[(size_t)t * (BB * HH) + gb * HH + u] = (__bf16)h;
        }
        __syncthreads();
    }
}

// ---------------------------------------------------------------------------
// Persistent attention decoder: one workgroup (512 thr = 16 waves), 64 steps.
// LDS (bf16): A0[128][304] xcat0 = [y|ct|pad|h0], A1[128][272] = [h0|h1],
//             G[128][520] gate/q/ctx scratch.   (~274KB)
// ---------------------------------------------------------------------------
#define A0S 304
#define A1S 272
#define GS  520

__device__ __forceinline__ void block_gemm_dec(const __bf16* A, int lda,
                                               const __bf16* W, int ldw,
                                               const float* bias, int K, int N,
                                               __bf16* Gout, int ldg, int gofs) {
    const int lane = threadIdx.x & 31;
    const int wave = threadIdx.x >> 5;
    const int ntiles = 8 * (N >> 4);
    for (int tile = wave; tile < ntiles; tile += 16) {
        const int m0 = (tile & 7) << 4;
        const int n0 = (tile >> 3) << 4;
        v8f acc = {};
        for (int k0 = 0; k0 < K; k0 += 32) {
            v16bf a = load_fragA(A, lda, m0, k0);
            v16bf b = load_fragB(W, ldw, n0, k0);
            acc = wmma_bf16(a, b, acc);
        }
        const int col = n0 + (lane & 15);
        const int rb  = m0 + ((lane >> 4) << 3);
        const float bv = bias[col];
#pragma unroll
        for (int i = 0; i < 8; ++i)
            Gout[(size_t)(rb + i) * ldg + gofs + col] = (__bf16)(acc[i] + bv);
    }
}

__global__ void decoder_k(const __bf16* __restrict__ wcat0, const float* __restrict__ b0,
                          const __bf16* __restrict__ wcat1, const float* __restrict__ b1,
                          const __bf16* __restrict__ Wq, const float* __restrict__ bq,
                          const __bf16* __restrict__ Wo, const float* __restrict__ bo,
                          const float* __restrict__ linW, const float* __restrict__ linb,
                          const float* __restrict__ kbuf, const float* __restrict__ vbuf,
                          float* __restrict__ out) {
    extern __shared__ char smem[];
    __bf16* A0 = (__bf16*)smem;                                  // [128][304]
    __bf16* A1 = (__bf16*)(smem + 128 * A0S * 2);                // [128][272]
    __bf16* G  = (__bf16*)(smem + 128 * A0S * 2 + 128 * A1S * 2);// [128][520]
    const int tid = threadIdx.x;
    const int total = 128 * (A0S + A1S + GS);
    for (int i = tid; i < total; i += 512) ((__bf16*)smem)[i] = (__bf16)0.f;
    float c0r[32], c1r[32];
#pragma unroll
    for (int j = 0; j < 32; ++j) { c0r[j] = 0.f; c1r[j] = 0.f; }
    const int gb  = tid >> 2;
    const int gu0 = (tid & 3) * 32;
    const int ab  = tid >> 2;       // batch for attention
    const int ah  = tid & 3;        // head
    __syncthreads();

    for (int step = 0; step < MAXL; ++step) {
        // layer0 gates: [y|ct|0|h0] @ [Wih0|0|Whh0]^T + b0
        block_gemm_dec(A0, A0S, wcat0, K0CAT, b0, K0CAT, G4, G, GS, 0);
        __syncthreads();
#pragma unroll
        for (int j = 0; j < 32; ++j) {
            const int u = gu0 + j;
            float gi = (float)G[gb * GS + u];
            float gf = (float)G[gb * GS + u + 128];
            float gg = (float)G[gb * GS + u + 256];
            float go = (float)G[gb * GS + u + 384];
            c0r[j] = sigmoidf_(gf) * c0r[j] + sigmoidf_(gi) * tanhf(gg);
            float h = sigmoidf_(go) * tanhf(c0r[j]);
            A1[gb * A1S + u] = (__bf16)h;           // h0 for layer1 GEMM
            A0[gb * A0S + 160 + u] = (__bf16)h;     // h0 for next step's layer0
        }
        __syncthreads();
        // layer1 gates: [h0|h1] @ [Wih1|Whh1]^T + b1
        block_gemm_dec(A1, A1S, wcat1, K1CAT, b1, K1CAT, G4, G, GS, 0);
        __syncthreads();
#pragma unroll
        for (int j = 0; j < 32; ++j) {
            const int u = gu0 + j;
            float gi = (float)G[gb * GS + u];
            float gf = (float)G[gb * GS + u + 128];
            float gg = (float)G[gb * GS + u + 256];
            float go = (float)G[gb * GS + u + 384];
            c1r[j] = sigmoidf_(gf) * c1r[j] + sigmoidf_(gi) * tanhf(gg);
            float h = sigmoidf_(go) * tanhf(c1r[j]);
            A1[gb * A1S + 128 + u] = (__bf16)h;     // h1
        }
        __syncthreads();
        // q = h1 @ Wq^T + bq  -> G cols [0,128)
        block_gemm_dec(A1 + 128, A1S, Wq, HH, bq, HH, HH, G, GS, 0);
        __syncthreads();
        // attention: one (b,head) per thread, online softmax over T (k/v L2-hot)
        {
            float qv[32];
#pragma unroll
            for (int d = 0; d < 32; ++d) qv[d] = (float)G[ab * GS + ah * HD + d];
            float m = -1e30f, l = 0.f, acc[32];
#pragma unroll
            for (int d = 0; d < 32; ++d) acc[d] = 0.f;
            for (int t = 0; t < TT; ++t) {
                const float* kp = kbuf + ((size_t)t * BB + ab) * HH + ah * HD;
                const float* vp = vbuf + ((size_t)t * BB + ab) * HH + ah * HD;
                if (t + 1 < TT) {
                    __builtin_prefetch(kp + BB * HH, 0, 1);
                    __builtin_prefetch(vp + BB * HH, 0, 1);
                }
                float s = 0.f;
#pragma unroll
                for (int d = 0; d < 32; ++d) s += qv[d] * kp[d];
                s *= 0.17677669529663687f; // 1/sqrt(32)
                float nm   = fmaxf(m, s);
                float corr = __expf(m - nm);
                float w    = __expf(s - nm);
                l = l * corr + w;
#pragma unroll
                for (int d = 0; d < 32; ++d) acc[d] = acc[d] * corr + w * vp[d];
                m = nm;
            }
            float inv = 1.f / l;
#pragma unroll
            for (int d = 0; d < 32; ++d)
                G[ab * GS + 256 + ah * HD + d] = (__bf16)(acc[d] * inv);
        }
        __syncthreads();
        // ct = ctx @ Wo^T + bo  -> A0 cols [4,132)
        block_gemm_dec(G + 256, GS, Wo, HH, bo, HH, HH, A0, A0S, 4);
        __syncthreads();
        // y = [h1|ct] @ linW^T + linb ; write output + feed back
        {
            const int b = tid >> 2, vi = tid & 3;
            float s = linb[vi];
#pragma unroll 8
            for (int j = 0; j < 128; ++j) s += (float)A1[b * A1S + 128 + j] * linW[vi * 256 + j];
#pragma unroll 8
            for (int j = 0; j < 128; ++j) s += (float)A0[b * A0S + 4 + j] * linW[vi * 256 + 128 + j];
            out[(size_t)step * (BB * VV) + b * VV + vi] = s;
            A0[b * A0S + vi] = (__bf16)s;
        }
        __syncthreads();
    }
}

// ---------------------------------------------------------------------------
// Host
// ---------------------------------------------------------------------------
extern "C" void kernel_launch(void* const* d_in, const int* in_sizes, int n_in,
                              void* d_out, int out_size, void* d_ws, size_t ws_size,
                              hipStream_t stream) {
    const float* x        = (const float*)d_in[0];
    const float* eWih0    = (const float*)d_in[1];
    const float* eWhh0    = (const float*)d_in[2];
    const float* eb0      = (const float*)d_in[3];
    const float* eWih1    = (const float*)d_in[4];
    const float* eWhh1    = (const float*)d_in[5];
    const float* eb1      = (const float*)d_in[6];
    const float* dWih0    = (const float*)d_in[7];
    const float* dWhh0    = (const float*)d_in[8];
    const float* db0      = (const float*)d_in[9];
    const float* dWih1    = (const float*)d_in[10];
    const float* dWhh1    = (const float*)d_in[11];
    const float* db1      = (const float*)d_in[12];
    const float* Wq       = (const float*)d_in[13];
    const float* bq       = (const float*)d_in[14];
    const float* Wk       = (const float*)d_in[15];
    const float* bk       = (const float*)d_in[16];
    const float* Wv       = (const float*)d_in[17];
    const float* bv       = (const float*)d_in[18];
    const float* Wo       = (const float*)d_in[19];
    const float* bo       = (const float*)d_in[20];
    const float* linW     = (const float*)d_in[21];
    const float* linb     = (const float*)d_in[22];
    float* out = (float*)d_out;

    char* ws = (char*)d_ws;
    size_t off = 0;
    auto alloc = [&](size_t bytes) -> char* {
        char* p = ws + off;
        off += (bytes + 255) & ~(size_t)255;
        return p;
    };
    __bf16* x_bf    = (__bf16*)alloc((size_t)TT * BB * DIN * 2);
    __bf16* wih0_bf = (__bf16*)alloc((size_t)G4 * DIN * 2);
    __bf16* whh0_bf = (__bf16*)alloc((size_t)G4 * HH * 2);
    __bf16* wih1_bf = (__bf16*)alloc((size_t)G4 * HH * 2);
    __bf16* whh1_bf = (__bf16*)alloc((size_t)G4 * HH * 2);
    __bf16* wk_bf   = (__bf16*)alloc((size_t)HH * HH * 2);
    __bf16* wv_bf   = (__bf16*)alloc((size_t)HH * HH * 2);
    __bf16* wq_bf   = (__bf16*)alloc((size_t)HH * HH * 2);
    __bf16* wo_bf   = (__bf16*)alloc((size_t)HH * HH * 2);
    __bf16* wcat0   = (__bf16*)alloc((size_t)G4 * K0CAT * 2);
    __bf16* wcat1   = (__bf16*)alloc((size_t)G4 * K1CAT * 2);
    float*  pre     = (float*) alloc((size_t)TT * BB * G4 * 4);   // reused layer0/1
    __bf16* hs0_bf  = (__bf16*)alloc((size_t)TT * BB * HH * 2);
    __bf16* enc_bf  = (__bf16*)alloc((size_t)TT * BB * HH * 2);
    float*  kbuf    = (float*) alloc((size_t)TT * BB * HH * 4);
    float*  vbuf    = (float*) alloc((size_t)TT * BB * HH * 4);

    // ---- precision conversion / weight packing ----
    auto cvt = [&](const float* s, __bf16* d, long long n) {
        int blocks = (int)((n + 256LL * 8 - 1) / (256LL * 8));
        if (blocks < 1) blocks = 1;
        if (blocks > 16384) blocks = 16384;
        f32_to_bf16_k<<<dim3(blocks), dim3(256), 0, stream>>>(s, d, n);
    };
    cvt(x,     x_bf,    (long long)TT * BB * DIN);
    cvt(eWih0, wih0_bf, (long long)G4 * DIN);
    cvt(eWhh0, whh0_bf, (long long)G4 * HH);
    cvt(eWih1, wih1_bf, (long long)G4 * HH);
    cvt(eWhh1, whh1_bf, (long long)G4 * HH);
    cvt(Wk,    wk_bf,   (long long)HH * HH);
    cvt(Wv,    wv_bf,   (long long)HH * HH);
    cvt(Wq,    wq_bf,   (long long)HH * HH);
    cvt(Wo,    wo_bf,   (long long)HH * HH);
    build_wcat0_k<<<dim3(288), dim3(256), 0, stream>>>(dWih0, dWhh0, wcat0);
    build_wcat1_k<<<dim3(256), dim3(256), 0, stream>>>(dWih1, dWhh1, wcat1);

    const int M = TT * BB;                 // 65536
    const size_t scan_smem = (size_t)128 * HLDS * 2 + (size_t)128 * GLDS * 4;
    const size_t dec_smem  = (size_t)128 * (A0S + A1S + GS) * 2;

    // ---- encoder layer 0 ----
    wmma_gemm_k<<<dim3(M / 64, G4 / 128), dim3(256), 0, stream>>>(
        x_bf, DIN, wih0_bf, DIN, eb0, pre, G4, DIN);
    lstm_scan_k<<<dim3(1), dim3(512), scan_smem, stream>>>(pre, whh0_bf, hs0_bf, TT);

    // ---- encoder layer 1 ----
    wmma_gemm_k<<<dim3(M / 64, G4 / 128), dim3(256), 0, stream>>>(
        hs0_bf, HH, wih1_bf, HH, eb1, pre, G4, HH);
    lstm_scan_k<<<dim3(1), dim3(512), scan_smem, stream>>>(pre, whh1_bf, enc_bf, TT);

    // ---- K/V projections ----
    wmma_gemm_k<<<dim3(M / 64, 1), dim3(256), 0, stream>>>(
        enc_bf, HH, wk_bf, HH, bk, kbuf, HH, HH);
    wmma_gemm_k<<<dim3(M / 64, 1), dim3(256), 0, stream>>>(
        enc_bf, HH, wv_bf, HH, bv, vbuf, HH, HH);

    // ---- attention decoder ----
    decoder_k<<<dim3(1), dim3(512), dec_smem, stream>>>(
        wcat0, db0, wcat1, db1, wq_bf, bq, wo_bf, bo, linW, linb, kbuf, vbuf, out);
}
